// TTMiniMaxM1SparseMoeBlock_1657857376609
// MI455X (gfx1250) — matrix operand order repaired and checked by reading, not compiled, not run on hardware
//
#include <hip/hip_runtime.h>
#include <hip/hip_bf16.h>
#include <cstdint>

// Problem sizes (fixed by the reference)
#define Hh 1024
#define Ff 2816
#define Ee 8
#define Tt 4096   // B*S tokens
#define TOPK 2

#define BF 128          // F-chunk per outer iteration (2816 = 22*128)
#define HS_STRIDE 136   // padded LDS stride for hidden tile (bank-conflict avoidance)

#if __has_builtin(__builtin_amdgcn_global_load_async_to_lds_b128)
#define HAS_ASYNC_LDS 1
#else
#define HAS_ASYNC_LDS 0
#endif

typedef __attribute__((ext_vector_type(16))) __bf16 v16bf;
typedef __attribute__((ext_vector_type(8)))  float  v8f;
typedef __attribute__((__vector_size__(4 * sizeof(int)))) int v4i_;
typedef __attribute__((address_space(1))) v4i_* gv4i_p;   // global 128-bit
typedef __attribute__((address_space(3))) v4i_* lv4i_p;   // LDS 128-bit

union BF16Frag { v16bf v; uint4 q[2]; };

__device__ __forceinline__ v8f vzero8() {
  v8f z = {0.f,0.f,0.f,0.f,0.f,0.f,0.f,0.f};
  return z;
}

// round-to-nearest-even fp32 -> bf16
__device__ __forceinline__ unsigned short f2bf(float f) {
  unsigned int u = __float_as_uint(f);
  unsigned int r = (u + 0x7fffu + ((u >> 16) & 1u)) >> 16;
  return (unsigned short)r;
}

__device__ __forceinline__ void cvt_bf16(const float* __restrict__ src,
                                         unsigned short* __restrict__ dst,
                                         size_t n4, size_t tid, size_t nth) {
  for (size_t i = tid; i < n4; i += nth) {
    float4 f = ((const float4*)src)[i];
    ushort4 o;
    o.x = f2bf(f.x); o.y = f2bf(f.y); o.z = f2bf(f.z); o.w = f2bf(f.w);
    ((ushort4*)dst)[i] = o;
  }
}

// ---------------------------------------------------------------------------
// K1: zero output + counters, convert activations/weights to bf16 workspace
// ---------------------------------------------------------------------------
__global__ __launch_bounds__(256) void prep_kernel(
    const float* __restrict__ x,  const float* __restrict__ w1,
    const float* __restrict__ w3, const float* __restrict__ w2,
    float* __restrict__ out,
    unsigned short* __restrict__ xb,  unsigned short* __restrict__ w1b,
    unsigned short* __restrict__ w3b, unsigned short* __restrict__ w2b,
    int* __restrict__ counts) {
  size_t tid = (size_t)blockIdx.x * blockDim.x + threadIdx.x;
  size_t nth = (size_t)gridDim.x * blockDim.x;
  if (tid < Ee) counts[tid] = 0;

  const size_t NOUT4 = (size_t)Tt * Hh / 4;
  float4 z = make_float4(0.f, 0.f, 0.f, 0.f);
  for (size_t i = tid; i < NOUT4; i += nth) ((float4*)out)[i] = z;

  cvt_bf16(x,  xb,  (size_t)Tt * Hh / 4,      tid, nth);
  cvt_bf16(w1, w1b, (size_t)Ee * Ff * Hh / 4, tid, nth);
  cvt_bf16(w3, w3b, (size_t)Ee * Ff * Hh / 4, tid, nth);
  cvt_bf16(w2, w2b, (size_t)Ee * Hh * Ff / 4, tid, nth);
}

// ---------------------------------------------------------------------------
// K2: router: logits (raw, to d_out tail), top-2 renormalized weights,
//     per-expert token scatter lists (atomic counters; order-independent).
//     256 threads = 32 tokens x 8 partial-K threads.
// ---------------------------------------------------------------------------
__global__ __launch_bounds__(256) void router_kernel(
    const float* __restrict__ x, const float* __restrict__ gw,
    float* __restrict__ logits, int* __restrict__ counts,
    int* __restrict__ tok_list, float* __restrict__ wgt_list) {
  __shared__ float gs[Ee * Hh];
  __shared__ float ps[256 * Ee];
  const int tid = threadIdx.x;
  for (int i = tid; i < Ee * Hh; i += 256) gs[i] = gw[i];
  __syncthreads();

  const int tbase = blockIdx.x * 32;
  const int sub   = tid & 7;     // K-slice (128 of H)
  const int trow  = tid >> 3;    // token in tile
  const float* xp = x + (size_t)(tbase + trow) * Hh + sub * 128;

  float acc[Ee];
#pragma unroll
  for (int e = 0; e < Ee; ++e) acc[e] = 0.f;
  for (int i = 0; i < 128; ++i) {
    float xv = xp[i];
    int h = sub * 128 + i;
#pragma unroll
    for (int e = 0; e < Ee; ++e) acc[e] += xv * gs[e * Hh + h];
  }
#pragma unroll
  for (int e = 0; e < Ee; ++e) ps[tid * Ee + e] = acc[e];
  __syncthreads();

  if (tid < 32) {
    const int t = tbase + tid;
    float l[Ee];
#pragma unroll
    for (int e = 0; e < Ee; ++e) {
      float s = 0.f;
      for (int j = 0; j < 8; ++j) s += ps[(tid * 8 + j) * Ee + e];
      l[e] = s;
      logits[(size_t)t * Ee + e] = s;     // raw router logits (2nd output)
    }
    // top-2 (first occurrence wins ties, matching lax.top_k)
    int e0 = 0;
#pragma unroll
    for (int e = 1; e < Ee; ++e) if (l[e] > l[e0]) e0 = e;
    int e1 = -1;
#pragma unroll
    for (int e = 0; e < Ee; ++e) {
      if (e == e0) continue;
      if (e1 < 0 || l[e] > l[e1]) e1 = e;
    }
    // renormalized softmax over the top-2: w0 = 1/(1+exp(l1-l0))
    float r   = __expf(l[e1] - l[e0]);
    float inv = 1.f / (1.f + r);
    int p0 = atomicAdd(&counts[e0], 1);
    tok_list[e0 * Tt + p0] = t; wgt_list[e0 * Tt + p0] = inv;
    int p1 = atomicAdd(&counts[e1], 1);
    tok_list[e1 * Tt + p1] = t; wgt_list[e1 * Tt + p1] = r * inv;
  }
}

// ---------------------------------------------------------------------------
// K3: grouped expert MLP. grid = (128 tiles, 8 experts); 256 threads = 8 waves.
//     Workgroup owns 32 tokens of one expert; F swept in 128-chunks:
//       phase1: hidden = silu(x*w1^T) * (x*w3^T)   (WMMA bf16, LDS x-tile)
//       phase2: out   += hidden * w2-slice         (WMMA bf16, LDS hidden)
//     Final: rows scaled by combine weight, f32 atomic-add into d_out.
// ---------------------------------------------------------------------------
__global__ __launch_bounds__(256) void moe_kernel(
    const unsigned short* __restrict__ xb,
    const unsigned short* __restrict__ w1b,
    const unsigned short* __restrict__ w3b,
    const unsigned short* __restrict__ w2b,
    const int* __restrict__ counts,
    const int* __restrict__ tok_list,
    const float* __restrict__ wgt_list,
    float* __restrict__ out) {
  extern __shared__ unsigned short smem[];
  unsigned short* xs = smem;                       // 32 x 1024 bf16 x-tile
  unsigned short* hs = smem + 32 * Hh;             // 32 x HS_STRIDE bf16 hidden
  int*   toks = (int*)(smem + 32 * Hh + 32 * HS_STRIDE);
  float* tokw = (float*)(toks + 32);

  const int e     = blockIdx.y;
  const int tbase = blockIdx.x * 32;
  const int n_e   = counts[e];
  if (tbase >= n_e) return;

  const int tid    = threadIdx.x;
  const int wave   = tid >> 5;
  const int lane   = tid & 31;
  const int lane16 = lane & 15;
  const int hi     = lane >> 4;     // lane half selects K sub-range

  if (tid < 32) {
    int gi    = tbase + tid;
    bool ok   = gi < n_e;
    int  idx  = ok ? gi : tbase;    // pad partial tile with a valid row
    toks[tid] = tok_list[e * Tt + idx];
    tokw[tid] = ok ? wgt_list[e * Tt + gi] : 0.f;
  }
  __syncthreads();

  // gather 32 token rows of bf16 x into LDS (128 bits = 8 bf16 per transfer).
  // Uniform trip count (16 iters * 256 threads) -> EXEC all-ones per lane,
  // so the async cache->LDS path is legal; ASYNCcnt tracks completion.
  for (int i = tid; i < 32 * (Hh / 8); i += 256) {
    int row = i >> 7, ch = i & 127;
    const unsigned short* src = xb + (size_t)toks[row] * Hh + ch * 8;
    unsigned short*       dst = xs + row * Hh + ch * 8;
#if HAS_ASYNC_LDS
    __builtin_amdgcn_global_load_async_to_lds_b128((gv4i_p)src, (lv4i_p)dst, 0, 0);
#else
    *(uint4*)dst = *(const uint4*)src;
#endif
  }
#if HAS_ASYNC_LDS
#if __has_builtin(__builtin_amdgcn_s_wait_asynccnt)
  __builtin_amdgcn_s_wait_asynccnt(0);
#else
  asm volatile("s_wait_asynccnt 0" ::: "memory");
#endif
#endif
  __syncthreads();

  v8f acc[8][2];
#pragma unroll
  for (int i = 0; i < 8; ++i) { acc[i][0] = vzero8(); acc[i][1] = vzero8(); }

  const size_t wb1  = (size_t)e * Ff * Hh;
  const int    fcol = wave * 16 + lane16;

  for (int fb = 0; fb < Ff; fb += BF) {
    // ---- phase 1: this wave computes hidden[32 x 16] for f = fb + wave*16
    v8f c10 = vzero8(), c11 = vzero8(), c30 = vzero8(), c31 = vzero8();
    const unsigned short* w1p = w1b + wb1 + (size_t)(fb + fcol) * Hh;
    const unsigned short* w3p = w3b + wb1 + (size_t)(fb + fcol) * Hh;
    if (fb + BF < Ff) {          // keep next F-block's weight columns warm
      __builtin_prefetch(w1p + (size_t)BF * Hh, 0, 3);
      __builtin_prefetch(w3p + (size_t)BF * Hh, 0, 3);
    }
    for (int k0 = 0; k0 < Hh; k0 += 32) {
      BF16Frag a0, a1, b1, b3;
      // A (16x32 bf16): lane half 0 -> K {0..7,16..23}, half 1 -> {8..15,24..31}
      const unsigned short* ap0 = xs + lane16 * Hh + k0 + hi * 8;
      a0.q[0] = *(const uint4*)(ap0);
      a0.q[1] = *(const uint4*)(ap0 + 16);
      const unsigned short* ap1 = ap0 + 16 * Hh;     // token rows 16..31
      a1.q[0] = *(const uint4*)(ap1);
      a1.q[1] = *(const uint4*)(ap1 + 16);
      // B (32x16 bf16): lane = column; lane half 0 -> K {0..7,16..23}, half 1 -> {8..15,24..31}
      const unsigned short* bp1 = w1p + k0 + hi * 8;
      b1.q[0] = *(const uint4*)(bp1);
      b1.q[1] = *(const uint4*)(bp1 + 16);
      const unsigned short* bp3 = w3p + k0 + hi * 8;
      b3.q[0] = *(const uint4*)(bp3);
      b3.q[1] = *(const uint4*)(bp3 + 16);
      c10 = __builtin_amdgcn_wmma_f32_16x16x32_bf16(false, a0.v, false, b1.v, (short)0, c10, false, false);
      c11 = __builtin_amdgcn_wmma_f32_16x16x32_bf16(false, a1.v, false, b1.v, (short)0, c11, false, false);
      c30 = __builtin_amdgcn_wmma_f32_16x16x32_bf16(false, a0.v, false, b3.v, (short)0, c30, false, false);
      c31 = __builtin_amdgcn_wmma_f32_16x16x32_bf16(false, a1.v, false, b3.v, (short)0, c31, false, false);
    }
    // silu(gate) * up  -> bf16 hidden tile in LDS
#pragma unroll
    for (int m = 0; m < 2; ++m) {
      v8f g = m ? c11 : c10;
      v8f u = m ? c31 : c30;
#pragma unroll
      for (int r = 0; r < 8; ++r) {
        float gv = g[r];
        float hv = (gv / (1.f + __expf(-gv))) * u[r];
        hs[(m * 16 + r + hi * 8) * HS_STRIDE + wave * 16 + lane16] = f2bf(hv);
      }
    }
    __syncthreads();

    // ---- phase 2: out[:, wave*128 .. +128] += hidden * w2-slice
    const size_t w2row = ((size_t)e * Hh + wave * 128) * Ff;
#pragma unroll
    for (int kk = 0; kk < BF; kk += 32) {
      BF16Frag a0, a1;
      const unsigned short* hp0 = hs + lane16 * HS_STRIDE + kk + hi * 8;
      a0.q[0] = *(const uint4*)(hp0);
      a0.q[1] = *(const uint4*)(hp0 + 16);
      const unsigned short* hp1 = hp0 + 16 * HS_STRIDE;
      a1.q[0] = *(const uint4*)(hp1);
      a1.q[1] = *(const uint4*)(hp1 + 16);
#pragma unroll
      for (int ht = 0; ht < 8; ++ht) {
        const unsigned short* bp =
            w2b + w2row + (size_t)(ht * 16 + lane16) * Ff + fb + kk + hi * 8;
        BF16Frag b2;
        b2.q[0] = *(const uint4*)(bp);
        b2.q[1] = *(const uint4*)(bp + 16);
        acc[ht][0] = __builtin_amdgcn_wmma_f32_16x16x32_bf16(false, a0.v, false, b2.v, (short)0, acc[ht][0], false, false);
        acc[ht][1] = __builtin_amdgcn_wmma_f32_16x16x32_bf16(false, a1.v, false, b2.v, (short)0, acc[ht][1], false, false);
      }
    }
    __syncthreads();
  }

  // ---- weighted scatter-combine (IEEE add is commutative -> deterministic)
#pragma unroll
  for (int ht = 0; ht < 8; ++ht) {
#pragma unroll
    for (int m = 0; m < 2; ++m) {
#pragma unroll
      for (int r = 0; r < 8; ++r) {
        int row = m * 16 + r + hi * 8;
        if (tbase + row < n_e) {
          int   t   = toks[row];
          float wv  = tokw[row];
          int   col = wave * 128 + ht * 16 + lane16;
          float v   = acc[ht][m][r] * wv;
          __hip_atomic_fetch_add(&out[(size_t)t * Hh + col], v,
                                 __ATOMIC_RELAXED, __HIP_MEMORY_SCOPE_AGENT);
        }
      }
    }
  }
}

// ---------------------------------------------------------------------------
extern "C" void kernel_launch(void* const* d_in, const int* in_sizes, int n_in,
                              void* d_out, int out_size, void* d_ws, size_t ws_size,
                              hipStream_t stream) {
  const float* x  = (const float*)d_in[0];   // [T, H]
  const float* gw = (const float*)d_in[1];   // [E, H]
  const float* w1 = (const float*)d_in[2];   // [E, F, H]
  const float* w3 = (const float*)d_in[3];   // [E, F, H]
  const float* w2 = (const float*)d_in[4];   // [E, H, F]
  float* out    = (float*)d_out;             // [T, H] then [T, E] logits
  float* logits = out + (size_t)Tt * Hh;

  char* ws = (char*)d_ws;
  unsigned short* xb  = (unsigned short*)ws; ws += (size_t)Tt * Hh * 2;
  unsigned short* w1b = (unsigned short*)ws; ws += (size_t)Ee * Ff * Hh * 2;
  unsigned short* w3b = (unsigned short*)ws; ws += (size_t)Ee * Ff * Hh * 2;
  unsigned short* w2b = (unsigned short*)ws; ws += (size_t)Ee * Hh * Ff * 2;
  int*   counts   = (int*)ws;                ws += 64;
  int*   tok_list = (int*)ws;                ws += (size_t)Ee * Tt * 4;
  float* wgt_list = (float*)ws;              ws += (size_t)Ee * Tt * 4;

  prep_kernel<<<4096, 256, 0, stream>>>(x, w1, w3, w2, out, xb, w1b, w3b, w2b, counts);
  router_kernel<<<Tt / 32, 256, 0, stream>>>(x, gw, logits, counts, tok_list, wgt_list);

  dim3 grid(Tt / 32, Ee);   // max tiles per expert x experts; idle tiles exit
  size_t smem = (size_t)32 * Hh * 2 + (size_t)32 * HS_STRIDE * 2 + 32 * 4 + 32 * 4;
  moe_kernel<<<grid, 256, smem, stream>>>(xb, w1b, w3b, w2b, counts, tok_list,
                                          wgt_list, out);
}
